// TransMIL_8143257993460
// MI455X (gfx1250) — matrix operand order repaired and checked by reading, not compiled, not run on hardware
//
#include <hip/hip_runtime.h>
#include <hip/hip_bf16.h>

// ---------------------------------------------------------------------------
// TransMIL forward for gfx1250 (MI455X, CDNA5).
// All GEMMs: bf16 inputs, v_wmma_f32_16x16x32_bf16, fp32 accumulate.
// Reductions (LN / softmax / landmark means / pinv denom) in fp32.
// GEMM primitive: C[M,N] = A[M,K] (row-major bf16) x Bt[N,K]^T (row-major bf16)
// so both operands are K-contiguous -> WMMA fragments are aligned 16B loads.
// ---------------------------------------------------------------------------

typedef __bf16 bf16;
typedef __attribute__((ext_vector_type(16))) __bf16 v16bf;
typedef __attribute__((ext_vector_type(8)))  float  v8f;

#define DEV __device__ __forceinline__

union V16U { uint4 q[2]; v16bf v; };

// A fragment, 16x32 (MxK), row-major source.
// lane m = lane&15 ; khalf = lane>>4
// elems 0..7  = K k0 + 8*khalf + (0..7)
// elems 8..15 = K k0 + 16 + 8*khalf + (0..7)
DEV v16bf load_a_frag(const bf16* A, int lda, int row, int k0, int lane) {
  int m  = row + (lane & 15);
  int kh = (lane >> 4) << 3;
  const bf16* p = A + (size_t)m * lda + k0 + kh;
  V16U u;
  u.q[0] = *(const uint4*)(p);
  u.q[1] = *(const uint4*)(p + 16);
  return u.v;
}

// B fragment, 32x16 (KxN), from Bt (N x K row-major).
// lane n = lane&15 ; K base = k0 + 8*(lane>>4) ; elems 0..7 -> K base + j, elems 8..15 -> K base + 16 + (j-8)
DEV v16bf load_b_frag(const bf16* Bt, int ldb, int col, int k0, int lane) {
  int n  = col + (lane & 15);
  int kb = k0 + ((lane >> 4) << 3);
  const bf16* p = Bt + (size_t)n * ldb + kb;
  V16U u;
  u.q[0] = *(const uint4*)(p);
  u.q[1] = *(const uint4*)(p + 16);
  return u.v;
}

struct GemmP {
  const bf16* A; const bf16* B; float* C; bf16* Cbf; const float* bias;
  long long sA, sB, sC, sCbf;
  int lda, ldb, ldc, ldcbf;
  int M, N, K;
  int relu, scale_lim;     // cols < scale_lim get *scale (used for q * DH^-0.5)
  float scale;
};

// 256 threads = 8 wave32. Block tile 128x64; wave (wid&3, wid>>2) owns 32x32:
// 2x2 grid of 16x16 WMMA tiles, K stepped by 32.
__global__ __launch_bounds__(256) void k_gemm(GemmP p) {
  const bf16* A = p.A + (size_t)blockIdx.z * p.sA;
  const bf16* B = p.B + (size_t)blockIdx.z * p.sB;
  int lane = threadIdx.x & 31;
  int wid  = threadIdx.x >> 5;
  int wm = blockIdx.y * 128 + (wid & 3) * 32;
  int wn = blockIdx.x * 64  + (wid >> 2) * 32;
  v8f acc00 = {}, acc01 = {}, acc10 = {}, acc11 = {};
  for (int k0 = 0; k0 < p.K; k0 += 32) {
    v16bf a0 = load_a_frag(A, p.lda, wm,      k0, lane);
    v16bf a1 = load_a_frag(A, p.lda, wm + 16, k0, lane);
    v16bf b0 = load_b_frag(B, p.ldb, wn,      k0, lane);
    v16bf b1 = load_b_frag(B, p.ldb, wn + 16, k0, lane);
    acc00 = __builtin_amdgcn_wmma_f32_16x16x32_bf16(false, a0, false, b0, (short)0, acc00, false, false);
    acc01 = __builtin_amdgcn_wmma_f32_16x16x32_bf16(false, a0, false, b1, (short)0, acc01, false, false);
    acc10 = __builtin_amdgcn_wmma_f32_16x16x32_bf16(false, a1, false, b0, (short)0, acc10, false, false);
    acc11 = __builtin_amdgcn_wmma_f32_16x16x32_bf16(false, a1, false, b1, (short)0, acc11, false, false);
  }
  float* C   = p.C   ? p.C   + (size_t)blockIdx.z * p.sC   : nullptr;
  bf16*  Cbf = p.Cbf ? p.Cbf + (size_t)blockIdx.z * p.sCbf : nullptr;
  int colh = lane & 15;
  int rowh = (lane >> 4) * 8;       // C/D: VGPR r -> M=r (+8 for lanes 16-31)
  v8f accs[4] = {acc00, acc01, acc10, acc11};
  #pragma unroll
  for (int mt = 0; mt < 2; ++mt)
    #pragma unroll
    for (int nt = 0; nt < 2; ++nt) {
      v8f a = accs[mt * 2 + nt];
      int col = wn + nt * 16 + colh;
      float bv = p.bias ? p.bias[col] : 0.f;
      float sc = (col < p.scale_lim) ? p.scale : 1.f;
      #pragma unroll
      for (int r = 0; r < 8; ++r) {
        int row = wm + mt * 16 + rowh + r;
        if (row >= p.M) continue;
        float v = a[r] + bv;
        if (p.relu) v = fmaxf(v, 0.f);
        v *= sc;
        if (C)   C[(size_t)row * p.ldc + col] = v;
        if (Cbf) Cbf[(size_t)row * p.ldcbf + col] = (bf16)v;
      }
    }
}

// dst[c*ldd + r] = bf16(scale * (scale_ptr?scale_ptr[z]:1) * src[r*lds + c] + diag*(r==c))
// Tiled 32x32 via LDS; covers weight transposes, v^T, and the pinv (aI - X^T) terms.
template <typename T>
__global__ __launch_bounds__(256) void k_transpose(bf16* dst, const T* src,
    int R, int C, int lds, int ldd, float scale, float diag,
    const float* scale_ptr, long long sSrc, long long sDst)
{
  __shared__ float tile[32][33];
  src += (size_t)blockIdx.z * sSrc;
  dst += (size_t)blockIdx.z * sDst;
  float sc = scale * (scale_ptr ? scale_ptr[blockIdx.z] : 1.f);
  int r0 = blockIdx.y * 32, c0 = blockIdx.x * 32;
  int tx = threadIdx.x & 31, ty = threadIdx.x >> 5;
  #pragma unroll
  for (int i = 0; i < 4; ++i) {
    int r = r0 + ty + i * 8, c = c0 + tx;
    if (r < R && c < C) tile[ty + i * 8][tx] = (float)src[(size_t)r * lds + c];
  }
  __syncthreads();
  #pragma unroll
  for (int i = 0; i < 4; ++i) {
    int r = r0 + tx, c = c0 + ty + i * 8;
    if (r < R && c < C)
      dst[(size_t)c * ldd + r] = (bf16)(sc * tile[tx][ty + i * 8] + ((r == c) ? diag : 0.f));
  }
}

__global__ __launch_bounds__(256) void k_convert(bf16* dst, const float* src, long long n,
    float scale, const float* scale_ptr, long long batch_elems)
{
  long long i = (long long)blockIdx.x * 256 + threadIdx.x;
  if (i >= n) return;
  float sc = scale;
  if (scale_ptr) sc *= scale_ptr[i / batch_elems];
  dst[i] = (bf16)(sc * src[i]);
}

__global__ void k_zero_bf(bf16* p, long long n) {
  long long i = (long long)blockIdx.x * 256 + threadIdx.x;
  if (i < n) p[i] = (bf16)0.f;
}

DEV float block_reduce_sum(float v, float* red) {
  int t = threadIdx.x;
  __syncthreads();
  red[t] = v; __syncthreads();
  for (int s = 128; s > 0; s >>= 1) { if (t < s) red[t] += red[t + s]; __syncthreads(); }
  return red[0];
}
DEV float block_reduce_max(float v, float* red) {
  int t = threadIdx.x;
  __syncthreads();
  red[t] = v; __syncthreads();
  for (int s = 128; s > 0; s >>= 1) { if (t < s) red[t] = fmaxf(red[t], red[t + s]); __syncthreads(); }
  return red[0];
}

// Row LN over 512 cols; writes fp32 (residual source) and bf16 at row+rowoff
// (directly assembling the front-padded Nystrom input).
__global__ __launch_bounds__(256) void k_layernorm(const float* x, const float* g, const float* b,
    float* y, bf16* ybf, int rowoff)
{
  __shared__ float red[256];
  long long row = blockIdx.x;
  const float* xr = x + row * 512;
  int t = threadIdx.x;
  float x0 = xr[t], x1 = xr[t + 256];
  float mu = block_reduce_sum(x0 + x1, red) * (1.f / 512.f);
  float d0 = x0 - mu, d1 = x1 - mu;
  float var = block_reduce_sum(d0 * d0 + d1 * d1, red) * (1.f / 512.f);
  float rstd = rsqrtf(var + 1e-5f);
  float y0 = d0 * rstd * g[t] + b[t];
  float y1 = d1 * rstd * g[t + 256] + b[t + 256];
  y[row * 512 + t] = y0; y[row * 512 + t + 256] = y1;
  bf16* br = ybf + (row + rowoff) * 512;
  br[t] = (bf16)y0; br[t + 256] = (bf16)y1;
}

// One block per row, cols multiple of 256 (256 or 10240). In-place fp32 + bf16 out.
__global__ __launch_bounds__(256) void k_softmax(float* x, bf16* xbf, int cols) {
  __shared__ float red[256];
  long long row = blockIdx.x;
  float* xr = x + row * (long long)cols;
  bf16* br = xbf + row * (long long)cols;
  int t = threadIdx.x, nv = cols >> 8;
  float vals[40];
  float m = -1e30f;
  for (int i = 0; i < nv; ++i) { vals[i] = xr[t + i * 256]; m = fmaxf(m, vals[i]); }
  m = block_reduce_max(m, red);
  float s = 0.f;
  for (int i = 0; i < nv; ++i) { vals[i] = __expf(vals[i] - m); s += vals[i]; }
  s = block_reduce_sum(s, red);
  float inv = 1.f / s;
  for (int i = 0; i < nv; ++i) {
    float o = vals[i] * inv;
    xr[t + i * 256] = o; br[t + i * 256] = (bf16)o;
  }
}

// Landmark means over L consecutive rows: ql/kl (H,256,64) bf16.
__global__ __launch_bounds__(256) void k_landmarks(bf16* ql, bf16* kl, const bf16* qkv, int L) {
  int idx = blockIdx.x * 256 + threadIdx.x;   // h*16384 + m*64 + d
  int d = idx & 63, m = (idx >> 6) & 255, h = idx >> 14;
  const bf16* base = qkv + (size_t)(m * L) * 1536 + h * 64 + d;
  float sq = 0.f, sk = 0.f;
  for (int j = 0; j < L; ++j) {
    sq += (float)base[(size_t)j * 1536];
    sk += (float)base[(size_t)j * 1536 + 512];
  }
  float inv = 1.f / (float)L;
  ql[idx] = (bf16)(sq * inv); kl[idx] = (bf16)(sk * inv);
}

// 1/(max abs row sum * max abs col sum) per head, on softmaxed a2.
__global__ __launch_bounds__(256) void k_denom(const float* a2, float* invden) {
  __shared__ float red[256];
  const float* x = a2 + (size_t)blockIdx.x * 65536;
  int t = threadIdx.x;
  float rs = 0.f, cs = 0.f;
  for (int j = 0; j < 256; ++j) { rs += fabsf(x[t * 256 + j]); cs += fabsf(x[j * 256 + t]); }
  float mr = block_reduce_max(rs, red);
  float mc = block_reduce_max(cs, red);
  if (t == 0) invden[blockIdx.x] = 1.f / (mr * mc);
}

// Depthwise 33-tap conv along sequence on v (bf16 in qkv), added to attn out.
__global__ void k_resconv(float* out, const bf16* qkv, const float* w, int S) {
  long long i = (long long)blockIdx.x * 256 + threadIdx.x;
  if (i >= (long long)S * 512) return;
  int c = (int)(i & 511); long long s = i >> 9;
  int h = c >> 6;
  float acc = 0.f;
  for (int t = 0; t < 33; ++t) {
    long long sp = s + t - 16;
    if (sp >= 0 && sp < S) acc += w[h * 33 + t] * (float)qkv[sp * 1536 + 1024 + c];
  }
  out[i] += acc;
}

DEV float dwconv2d(const float* f, const float* wk, int K, int hh, int ww, int c) {
  int half = K >> 1; float a = 0.f;
  for (int i = 0; i < K; ++i) {
    int r = hh + i - half; if ((unsigned)r >= 100u) continue;
    for (int j = 0; j < K; ++j) {
      int cc = ww + j - half; if ((unsigned)cc >= 100u) continue;
      a += wk[i * K + j] * f[((size_t)(r * 100 + cc)) * 512 + c];
    }
  }
  return a;
}

__global__ void k_ppeg(float* dst, const float* src, const float* w7, const float* b7,
                       const float* w5, const float* b5, const float* w3, const float* b3)
{
  long long i = (long long)blockIdx.x * 256 + threadIdx.x;
  if (i >= 10000LL * 512) return;
  int c = (int)(i & 511); int pix = (int)(i >> 9);
  int hh = pix / 100, ww = pix % 100;
  const float* f = src + 512;   // feature rows start after cls
  float v = f[(size_t)pix * 512 + c];
  float acc = v
    + dwconv2d(f, w7 + c * 49, 7, hh, ww, c) + b7[c]
    + dwconv2d(f, w5 + c * 25, 5, hh, ww, c) + b5[c]
    + dwconv2d(f, w3 + c * 9,  3, hh, ww, c) + b3[c];
  dst[512 + (size_t)pix * 512 + c] = acc;
}

__global__ void k_slice_add(float* seq, const float* xln, const float* proj,
                            long long n, long long shift) {
  long long i = (long long)blockIdx.x * 256 + threadIdx.x;
  if (i < n) seq[i] = xln[i] + proj[i + shift];
}

__global__ void k_copy512(float* dst, const float* src) {
  dst[threadIdx.x] = src[threadIdx.x];
  dst[threadIdx.x + 256] = src[threadIdx.x + 256];
}

// Final LN of cls row + 512x2 classifier (too small for WMMA).
__global__ __launch_bounds__(256) void k_head(float* out, const float* x, const float* g,
    const float* b, const float* w, const float* bias)
{
  __shared__ float red[256];
  __shared__ float v[512];
  int t = threadIdx.x;
  float x0 = x[t], x1 = x[t + 256];
  float mu = block_reduce_sum(x0 + x1, red) * (1.f / 512.f);
  float d0 = x0 - mu, d1 = x1 - mu;
  float var = block_reduce_sum(d0 * d0 + d1 * d1, red) * (1.f / 512.f);
  float rstd = rsqrtf(var + 1e-5f);
  v[t] = d0 * rstd * g[t] + b[t];
  v[t + 256] = d1 * rstd * g[t + 256] + b[t + 256];
  __syncthreads();
  float p0 = v[t] * w[2 * t]     + v[t + 256] * w[2 * (t + 256)];
  float p1 = v[t] * w[2 * t + 1] + v[t + 256] * w[2 * (t + 256) + 1];
  p0 = block_reduce_sum(p0, red);
  p1 = block_reduce_sum(p1, red);
  if (t == 0) { out[0] = p0 + bias[0]; out[1] = p1 + bias[1]; }
}

// ---------------------------------------------------------------------------
extern "C" void kernel_launch(void* const* d_in, const int* in_sizes, int n_in,
                              void* d_out, int out_size, void* d_ws, size_t ws_size,
                              hipStream_t stream)
{
  (void)in_sizes; (void)n_in; (void)out_size; (void)ws_size;
  const float* h        = (const float*)d_in[0];
  const float* fc1_w    = (const float*)d_in[1];
  const float* fc1_b    = (const float*)d_in[2];
  const float* cls_tok  = (const float*)d_in[3];
  const float* nrm_g[2] = {(const float*)d_in[4],  (const float*)d_in[10]};
  const float* nrm_b[2] = {(const float*)d_in[5],  (const float*)d_in[11]};
  const float* qkv_w[2] = {(const float*)d_in[6],  (const float*)d_in[12]};
  const float* out_w[2] = {(const float*)d_in[7],  (const float*)d_in[13]};
  const float* out_b[2] = {(const float*)d_in[8],  (const float*)d_in[14]};
  const float* res_w[2] = {(const float*)d_in[9],  (const float*)d_in[15]};
  const float* pw7 = (const float*)d_in[16], *pb7 = (const float*)d_in[17];
  const float* pw5 = (const float*)d_in[18], *pb5 = (const float*)d_in[19];
  const float* pw3 = (const float*)d_in[20], *pb3 = (const float*)d_in[21];
  const float* fng = (const float*)d_in[22], *fnb = (const float*)d_in[23];
  const float* fc2_w = (const float*)d_in[24], *fc2_b = (const float*)d_in[25];

  const int N0 = 10000, F = 1024, D = 512, Hh = 8;
  const int S = 10240, NC = 10001, PAD = 239, L = 40, MP1 = 10112;

  size_t off = 0;
  char* basep = (char*)d_ws;
  auto alloc = [&](size_t bytes) -> void* {
    off = (off + 255) & ~(size_t)255;
    void* p = basep + off; off += bytes; return p;
  };
  auto fa = [&](size_t e) { return (float*)alloc(e * 4); };
  auto ba = [&](size_t e) { return (bf16*)alloc(e * 2); };

  bf16* fc1T    = ba((size_t)D * F);
  bf16* qkvT0   = ba((size_t)(3 * D) * D);
  bf16* qkvT1   = ba((size_t)(3 * D) * D);
  bf16* outT0   = ba((size_t)D * D);
  bf16* outT1   = ba((size_t)D * D);
  bf16* h_bf    = ba((size_t)MP1 * F);
  float* seqA   = fa((size_t)(MP1 + 1) * D);
  float* seqB   = fa((size_t)NC * D);
  float* xln    = fa((size_t)NC * D);
  bf16* xp_bf   = ba((size_t)S * D);
  bf16* qkv_bf  = ba((size_t)S * 3 * D);
  bf16* ql_bf   = ba((size_t)Hh * 256 * 64);
  bf16* kl_bf   = ba((size_t)Hh * 256 * 64);
  float* a2     = fa((size_t)Hh * 65536);
  bf16* a2_bf   = ba((size_t)Hh * 65536);
  float* invden = fa(Hh);
  bf16* z_bf    = ba((size_t)Hh * 65536);
  bf16* zT_bf   = ba((size_t)Hh * 65536);
  float* uf     = fa((size_t)Hh * 65536);
  bf16* u_bf    = ba((size_t)Hh * 65536);
  bf16* t1_bf   = ba((size_t)Hh * 65536);
  float* pf     = fa((size_t)Hh * 65536);
  float* af     = fa((size_t)S * 256);
  bf16* a1_bf   = ba((size_t)S * 256);
  bf16* a3_bf   = ba((size_t)256 * S);
  bf16* vT_bf   = ba((size_t)64 * S);
  float* a3v    = fa(256 * 64);
  bf16* a3vT    = ba(64 * 256);
  float* tmat   = fa(256 * 64);
  bf16* tT      = ba(64 * 256);
  float* attn   = fa((size_t)S * D);
  bf16* attn_bf = ba((size_t)S * D);
  float* projf  = fa((size_t)S * D);

  auto gemm = [&](const bf16* A, long long sA, int lda,
                  const bf16* B, long long sB, int ldb,
                  float* C, long long sC, int ldc,
                  bf16* Cbf, long long sCbf, int ldcbf,
                  const float* bias, int M, int N, int K,
                  int relu, int slim, float sc, int batch) {
    GemmP p;
    p.A = A; p.B = B; p.C = C; p.Cbf = Cbf; p.bias = bias;
    p.sA = sA; p.sB = sB; p.sC = sC; p.sCbf = sCbf;
    p.lda = lda; p.ldb = ldb; p.ldc = ldc; p.ldcbf = ldcbf;
    p.M = M; p.N = N; p.K = K; p.relu = relu; p.scale_lim = slim; p.scale = sc;
    dim3 g((unsigned)((N + 63) / 64), (unsigned)((M + 127) / 128), (unsigned)batch);
    k_gemm<<<g, 256, 0, stream>>>(p);
  };
  auto transpose_f = [&](bf16* dst, const float* src, int R, int C, int lds, int ldd,
                         float sc, float dg, const float* sp, long long sS, long long sD, int batch) {
    dim3 g((unsigned)((C + 31) / 32), (unsigned)((R + 31) / 32), (unsigned)batch);
    k_transpose<float><<<g, 256, 0, stream>>>(dst, src, R, C, lds, ldd, sc, dg, sp, sS, sD);
  };
  auto transpose_b = [&](bf16* dst, const bf16* src, int R, int C, int lds, int ldd) {
    dim3 g((unsigned)((C + 31) / 32), (unsigned)((R + 31) / 32), 1u);
    k_transpose<bf16><<<g, 256, 0, stream>>>(dst, src, R, C, lds, ldd, 1.f, 0.f, nullptr, 0, 0);
  };
  auto convert = [&](bf16* dst, const float* src, long long n, float sc,
                     const float* sp, long long be) {
    k_convert<<<dim3((unsigned)((n + 255) / 256)), 256, 0, stream>>>(dst, src, n, sc, sp, be);
  };

  // ---- weight prep (recomputed every call: launch must be stateless) ----
  transpose_f(fc1T,  fc1_w,    F, D,     D,     F, 1.f, 0.f, nullptr, 0, 0, 1);
  transpose_f(qkvT0, qkv_w[0], D, 3 * D, 3 * D, D, 1.f, 0.f, nullptr, 0, 0, 1);
  transpose_f(qkvT1, qkv_w[1], D, 3 * D, 3 * D, D, 1.f, 0.f, nullptr, 0, 0, 1);
  transpose_f(outT0, out_w[0], D, D,     D,     D, 1.f, 0.f, nullptr, 0, 0, 1);
  transpose_f(outT1, out_w[1], D, D,     D,     D, 1.f, 0.f, nullptr, 0, 0, 1);

  // ---- fc1 + relu, output into seqA rows 1..10000; cls token at row 0 ----
  convert(h_bf, h, (long long)N0 * F, 1.f, nullptr, 1);
  gemm(h_bf, 0, F, fc1T, 0, F, seqA + D, 0, D, nullptr, 0, 0,
       fc1_b, N0, D, F, /*relu*/1, 0, 1.f, 1);
  k_copy512<<<1, 256, 0, stream>>>(seqA, cls_tok);

  float* cur = seqA;
  for (int l = 0; l < 2; ++l) {
    bf16* qkvT = l ? qkvT1 : qkvT0;
    bf16* outT = l ? outT1 : outT0;

    // LN -> xln (fp32 residual) + xp_bf rows PAD.. (bf16), zero front pad
    k_layernorm<<<NC, 256, 0, stream>>>(cur, nrm_g[l], nrm_b[l], xln, xp_bf, PAD);
    k_zero_bf<<<(PAD * D + 255) / 256, 256, 0, stream>>>(xp_bf, (long long)PAD * D);

    // qkv GEMM, bf16 out with q-part pre-scaled by DH^-0.5
    gemm(xp_bf, 0, D, qkvT, 0, D, nullptr, 0, 0, qkv_bf, 0, 3 * D,
         nullptr, S, 3 * D, D, 0, D, 0.125f, 1);

    // landmarks + a2 + softmax + pinv denominator (all heads batched)
    k_landmarks<<<Hh * 256 * 64 / 256, 256, 0, stream>>>(ql_bf, kl_bf, qkv_bf, L);
    gemm(ql_bf, 16384, 64, kl_bf, 16384, 64, a2, 65536, 256, nullptr, 0, 0,
         nullptr, 256, 256, 64, 0, 0, 1.f, Hh);
    k_softmax<<<Hh * 256, 256, 0, stream>>>(a2, a2_bf, 256);
    k_denom<<<Hh, 256, 0, stream>>>(a2, invden);

    // Newton-Schulz pinv, batched over heads. z0 = x^T/denom.
    transpose_f(z_bf, a2, 256, 256, 256, 256, 1.f, 0.f, invden, 65536, 65536, Hh);
    convert(zT_bf, a2, (long long)Hh * 65536, 1.f, invden, 65536);
    for (int it = 0; it < 6; ++it) {
      gemm(a2_bf, 65536, 256, zT_bf, 65536, 256, uf, 65536, 256, nullptr, 0, 0,
           nullptr, 256, 256, 256, 0, 0, 1.f, Hh);                       // u = x z
      convert(u_bf, uf, (long long)Hh * 65536, 1.f, nullptr, 1);
      transpose_f(t1_bf, uf, 256, 256, 256, 256, -1.f, 7.f, nullptr, 65536, 65536, Hh);
      gemm(u_bf, 65536, 256, t1_bf, 65536, 256, pf, 65536, 256, nullptr, 0, 0,
           nullptr, 256, 256, 256, 0, 0, 1.f, Hh);                       // p = u(7I-u)
      transpose_f(t1_bf, pf, 256, 256, 256, 256, -1.f, 15.f, nullptr, 65536, 65536, Hh);
      gemm(u_bf, 65536, 256, t1_bf, 65536, 256, uf, 65536, 256, nullptr, 0, 0,
           nullptr, 256, 256, 256, 0, 0, 1.f, Hh);                       // r = u(15I-p)
      transpose_f(t1_bf, uf, 256, 256, 256, 256, -1.f, 13.f, nullptr, 65536, 65536, Hh);
      gemm(z_bf, 65536, 256, t1_bf, 65536, 256, pf, 65536, 256, nullptr, 0, 0,
           nullptr, 256, 256, 256, 0, 0, 1.f, Hh);                       // zn = z(13I-r)
      convert(z_bf, pf, (long long)Hh * 65536, 0.25f, nullptr, 1);
      transpose_f(zT_bf, pf, 256, 256, 256, 256, 0.25f, 0.f, nullptr, 65536, 65536, Hh);
    }

    // per-head attention streams (reuses the big S x 256 buffers)
    for (int hd = 0; hd < Hh; ++hd) {
      const bf16* qh = qkv_bf + hd * 64;
      const bf16* kh = qkv_bf + 512 + hd * 64;
      const bf16* vh = qkv_bf + 1024 + hd * 64;
      gemm(qh, 0, 3 * D, kl_bf + (size_t)hd * 16384, 0, 64, af, 0, 256,
           nullptr, 0, 0, nullptr, S, 256, 64, 0, 0, 1.f, 1);            // a1 logits
      k_softmax<<<S, 256, 0, stream>>>(af, a1_bf, 256);
      gemm(ql_bf + (size_t)hd * 16384, 0, 64, kh, 0, 3 * D, af, 0, S,
           nullptr, 0, 0, nullptr, 256, S, 64, 0, 0, 1.f, 1);            // a3 logits
      k_softmax<<<256, 256, 0, stream>>>(af, a3_bf, S);
      transpose_b(vT_bf, vh, S, 64, 3 * D, S);                           // v^T
      gemm(a3_bf, 0, S, vT_bf, 0, S, a3v, 0, 64, nullptr, 0, 0,
           nullptr, 256, 64, S, 0, 0, 1.f, 1);                           // a3 v
      transpose_f(a3vT, a3v, 256, 64, 64, 256, 1.f, 0.f, nullptr, 0, 0, 1);
      gemm(z_bf + (size_t)hd * 65536, 0, 256, a3vT, 0, 256, tmat, 0, 64,
           nullptr, 0, 0, nullptr, 256, 64, 256, 0, 0, 1.f, 1);          // t = pinv @ (a3 v)
      transpose_f(tT, tmat, 256, 64, 64, 256, 1.f, 0.f, nullptr, 0, 0, 1);
      gemm(a1_bf, 0, 256, tT, 0, 256, attn + hd * 64, 0, D,
           nullptr, 0, 0, nullptr, S, 64, 256, 0, 0, 1.f, 1);            // a1 @ t
    }

    k_resconv<<<(unsigned)(((long long)S * 512 + 255) / 256), 256, 0, stream>>>(
        attn, qkv_bf, res_w[l], S);
    convert(attn_bf, attn, (long long)S * D, 1.f, nullptr, 1);
    gemm(attn_bf, 0, D, outT, 0, D, projf, 0, D, nullptr, 0, 0,
         out_b[l], S, D, D, 0, 0, 1.f, 1);                               // out proj
    k_slice_add<<<(unsigned)(((long long)NC * D + 255) / 256), 256, 0, stream>>>(
        seqB, xln, projf, (long long)NC * D, (long long)PAD * D);        // residual + slice

    if (l == 0) {
      k_ppeg<<<(unsigned)((10000LL * 512 + 255) / 256), 256, 0, stream>>>(
          seqA, seqB, pw7, pb7, pw5, pb5, pw3, pb3);
      k_copy512<<<1, 256, 0, stream>>>(seqA, seqB);   // cls row passes through
      cur = seqA;
    } else {
      cur = seqB;
    }
  }

  k_head<<<1, 256, 0, stream>>>((float*)d_out, cur, fng, fnb, fc2_w, fc2_b);
}
